// RecurrentContentTransfer_63900523430660
// MI455X (gfx1250) — matrix operand order repaired and checked by reading, not compiled, not run on hardware
//
#include <hip/hip_runtime.h>
#include <hip/hip_bf16.h>

// ---------------------------------------------------------------------------
// Problem constants (from reference): B=16, C=256, CB=64, WIDTH=32, F=2048
// LSTM: 2 layers, hidden 2048, gates 4*2048=8192, combined K = 4096 (x||h)
// ---------------------------------------------------------------------------
#define BATCH   16
#define CHN     256
#define CB      64
#define WID     32
#define FEAT    2048            // CB*WID
#define GATES   8192            // 4*FEAT
#define KCAT    4096            // FEAT (x) + FEAT (h)
#define NT      512             // GATES/16 N-tiles
#define KT      128             // KCAT/32  K-tiles
#define STEPS   63              // 32 encode + 31 generate

typedef __bf16 bf16;
typedef __attribute__((ext_vector_type(16))) __bf16 v16bf;
typedef __attribute__((ext_vector_type(8)))  float  v8f;

// ---------------------------------------------------------------------------
// Pack W_ih (fp32 [8192][2048]) || W_hh (fp32 [8192][2048]) into bf16 tiles,
// pre-swizzled into a 32x16 B-fragment lane layout (this code's contiguous k order):
//   tile (nt,kt): lane l owns column n = nt*16 + (l&15),
//                 k-range = kt*32 + (l>>4)*16 + e, e = 0..15 (contiguous).
// Tile storage: 512 bf16, lane-major (lane*16 + e)  -> GEMM loads 2x b128/lane.
// ---------------------------------------------------------------------------
__global__ void pack_w(const float* __restrict__ Wih, const float* __restrict__ Whh,
                       bf16* __restrict__ Wp) {
    int idx  = blockIdx.x * 256 + threadIdx.x;     // < 512*128*512 = 33554432
    int e    = idx & 15;
    int lane = (idx >> 4) & 31;
    int kt   = (idx >> 9) & 127;
    int nt   = idx >> 16;
    int n    = nt * 16 + (lane & 15);
    int k    = kt * 32 + (lane >> 4) * 16 + e;
    float v  = (k < FEAT) ? Wih[(size_t)n * FEAT + k]
                          : Whh[(size_t)n * FEAT + (k - FEAT)];
    Wp[idx] = (bf16)v;
}

// ---------------------------------------------------------------------------
// 1x1 conv in:  bn[b][co][h][w] = sum_c x[b][c][h][w]*w_in[co][c] + b_in[co]
// stored as bf16 sequence rows: bnseq[h][b][w*64+co]
// ---------------------------------------------------------------------------
__global__ void conv_in(const float* __restrict__ x, const float* __restrict__ w_in,
                        const float* __restrict__ b_in, bf16* __restrict__ bnseq) {
    int idx = blockIdx.x * 256 + threadIdx.x;      // < 16*32*32*64 = 1048576
    int co  = idx & 63;
    int w   = (idx >> 6) & 31;
    int h   = (idx >> 11) & 31;
    int b   = idx >> 16;
    const float* xp = x + (size_t)b * (CHN * WID * WID) + h * WID + w;
    const float* wp = w_in + co * CHN;
    float acc = b_in[co];
    for (int c = 0; c < CHN; ++c) acc += xp[c * (WID * WID)] * wp[c];
    bnseq[((size_t)h * BATCH + b) * FEAT + w * CB + co] = (bf16)acc;
}

// ---------------------------------------------------------------------------
// LSTM gate GEMM for one layer-step:
//   gates[16][8192] = [Ax | Ah] (bf16 16x4096) * Wp^T + (b_ih + b_hh)
// Block = 4 waves, one N-tile (nt = blockIdx.x), split-K x4, LDS reduce.
// Each wave: 32 x v_wmma_f32_16x16x32_bf16, streaming 1KB weight tiles (L2).
// Each wave's K-chunk (1024 k's) lies entirely in Ax (waves 0,1) or Ah
// (waves 2,3): hoist the operand-half select out of the hot loop.
// ---------------------------------------------------------------------------
__global__ void __launch_bounds__(128)
lstm_gemm(const bf16* __restrict__ Ax, const bf16* __restrict__ Ah,
          const bf16* __restrict__ Wp, const float* __restrict__ b_ih,
          const float* __restrict__ b_hh, float* __restrict__ gates) {
    const int nt    = blockIdx.x;          // 0..511
    const int wv    = threadIdx.x >> 5;    // 0..3
    const int lane  = threadIdx.x & 31;
    const int khalf = lane >> 4;           // 0/1
    const int m     = lane & 15;           // A row

    union { v16bf v; uint4 u[2]; } afrag, bfrag;
    union { v8f   v; float f[8];  } acc;
#pragma unroll
    for (int r = 0; r < 8; ++r) acc.f[r] = 0.0f;

    const bf16* wtile = Wp + ((size_t)nt * KT + wv * 32) * 512 + lane * 16;

    // A fragment (contiguous k order, matching packed B):
    //   lanes 0-15 : elems 0-15 = k0..15
    //   lanes 16-31: elems 0-15 = k16..31
    const int kstart = wv * 32 * 32;       // first k of this wave's chunk
    const bf16* aptr = (kstart < FEAT)
        ? (Ax + (size_t)m * FEAT + kstart + khalf * 16)
        : (Ah + (size_t)m * FEAT + (kstart - FEAT) + khalf * 16);

#pragma unroll 4
    for (int i = 0; i < 32; ++i) {
        afrag.u[0] = *(const uint4*)(aptr);
        afrag.u[1] = *(const uint4*)(aptr + 8);
        bfrag.u[0] = *(const uint4*)(wtile);
        bfrag.u[1] = *(const uint4*)(wtile + 8);
        aptr  += 32;
        wtile += 512;
        acc.v = __builtin_amdgcn_wmma_f32_16x16x32_bf16(
            false, afrag.v, false, bfrag.v, (short)0, acc.v, false, false);
    }

    // split-K reduction across the 4 waves
    __shared__ float red[4][32][8];
#pragma unroll
    for (int r = 0; r < 8; ++r) red[wv][lane][r] = acc.f[r];
    __syncthreads();
    if (wv == 0) {
        const int n    = nt * 16 + (lane & 15);
        const int mtop = (lane >> 4) * 8;      // C/D layout: VGPR r -> M = r + 8*(lane>>4)
        const float bias = b_ih[n] + b_hh[n];
#pragma unroll
        for (int r = 0; r < 8; ++r) {
            float s = red[0][lane][r] + red[1][lane][r] +
                      red[2][lane][r] + red[3][lane][r];
            gates[(size_t)(mtop + r) * GATES + n] = s + bias;
        }
    }
}

// ---------------------------------------------------------------------------
// LSTM cell elementwise: c' = sig(f)*c + sig(i)*tanh(g); h' = sig(o)*tanh(c')
// Writes h' as bf16 (next GEMM A operand) and optionally fp32 sequence row.
// ---------------------------------------------------------------------------
__device__ __forceinline__ float sigf(float x) { return 1.0f / (1.0f + __expf(-x)); }

__global__ void lstm_cell(const float* __restrict__ gates, float* __restrict__ c,
                          bf16* __restrict__ hbf, float* __restrict__ seq_row) {
    int idx = blockIdx.x * 256 + threadIdx.x;  // < 16*2048 = 32768
    int m = idx >> 11, j = idx & (FEAT - 1);
    const float* g = gates + (size_t)m * GATES;
    float gi = g[j], gf = g[j + FEAT], gg = g[j + 2 * FEAT], go = g[j + 3 * FEAT];
    float cn = sigf(gf) * c[idx] + sigf(gi) * tanhf(gg);
    float hn = sigf(go) * tanhf(cn);
    c[idx]   = cn;
    hbf[idx] = (bf16)hn;
    if (seq_row) seq_row[idx] = hn;
}

// ---------------------------------------------------------------------------
// Output: out[b][ch][h][w] = x for h<32, else 1x1 conv of seqout row h-32.
// ---------------------------------------------------------------------------
__global__ void conv_out_concat(const float* __restrict__ x, const float* __restrict__ seqout,
                                const float* __restrict__ w_out, const float* __restrict__ b_out,
                                float* __restrict__ out) {
    int idx = blockIdx.x * 256 + threadIdx.x;  // < 16*256*64*32 = 8388608
    int w  = idx & 31;
    int h  = (idx >> 5) & 63;
    int ch = (idx >> 11) & 255;
    int b  = idx >> 19;
    if (h < WID) {
        out[idx] = x[(((size_t)b * CHN + ch) * WID + h) * WID + w];
    } else {
        const int s = h - WID;
        const float* row = seqout + ((size_t)s * BATCH + b) * FEAT + w * CB;
        const float* wo  = w_out + ch * CB;
        float acc = b_out[ch];
#pragma unroll
        for (int cc = 0; cc < CB; ++cc) acc += row[cc] * wo[cc];
        out[idx] = acc;
    }
}

// ---------------------------------------------------------------------------
extern "C" void kernel_launch(void* const* d_in, const int* in_sizes, int n_in,
                              void* d_out, int out_size, void* d_ws, size_t ws_size,
                              hipStream_t stream) {
    const float* x     = (const float*)d_in[0];
    const float* w_in  = (const float*)d_in[1];
    const float* b_in  = (const float*)d_in[2];
    const float* W_ih0 = (const float*)d_in[3];
    const float* W_hh0 = (const float*)d_in[4];
    const float* b_ih0 = (const float*)d_in[5];
    const float* b_hh0 = (const float*)d_in[6];
    const float* W_ih1 = (const float*)d_in[7];
    const float* W_hh1 = (const float*)d_in[8];
    const float* b_ih1 = (const float*)d_in[9];
    const float* b_hh1 = (const float*)d_in[10];
    const float* w_out = (const float*)d_in[11];
    const float* b_out = (const float*)d_in[12];
    float* out = (float*)d_out;

    // workspace carve-up (bf16 weights resident in 192MB L2 across all steps)
    char* ws = (char*)d_ws;
    size_t off = 0;
    auto carve = [&](size_t bytes) -> void* {
        void* p = ws + off;
        off = (off + bytes + 255) & ~(size_t)255;
        return p;
    };
    const size_t WPB = (size_t)GATES * KCAT * sizeof(bf16);      // 64 MB per layer
    bf16*  Wp0    = (bf16*)carve(WPB);
    bf16*  Wp1    = (bf16*)carve(WPB);
    bf16*  bnseq  = (bf16*)carve((size_t)WID * BATCH * FEAT * sizeof(bf16));
    float* seqout = (float*)carve((size_t)WID * BATCH * FEAT * sizeof(float));
    float* gates  = (float*)carve((size_t)BATCH * GATES * sizeof(float));
    bf16*  h0bf   = (bf16*)carve((size_t)BATCH * FEAT * sizeof(bf16));
    bf16*  h1bf   = (bf16*)carve((size_t)BATCH * FEAT * sizeof(bf16));
    float* c0     = (float*)carve((size_t)BATCH * FEAT * sizeof(float));
    float* c1     = (float*)carve((size_t)BATCH * FEAT * sizeof(float));
    if (off > ws_size) return;  // workspace too small: bail deterministically

    // one-time (per call) weight pack to bf16 WMMA-B tile layout
    pack_w<<<131072, 256, 0, stream>>>(W_ih0, W_hh0, Wp0);
    pack_w<<<131072, 256, 0, stream>>>(W_ih1, W_hh1, Wp1);

    // input bottleneck conv -> bf16 sequence rows
    conv_in<<<4096, 256, 0, stream>>>(x, w_in, b_in, bnseq);

    // zero initial LSTM state (graph-capture-safe memset nodes)
    hipMemsetAsync(h0bf, 0, (size_t)BATCH * FEAT * sizeof(bf16), stream);
    hipMemsetAsync(h1bf, 0, (size_t)BATCH * FEAT * sizeof(bf16), stream);
    hipMemsetAsync(c0,   0, (size_t)BATCH * FEAT * sizeof(float), stream);
    hipMemsetAsync(c1,   0, (size_t)BATCH * FEAT * sizeof(float), stream);

    // 63 sequential steps: 32 encode (input rows) + 31 autoregressive (h1 fed back)
    for (int s = 0; s < STEPS; ++s) {
        const bf16* Ax0 = (s < WID) ? (bnseq + (size_t)s * BATCH * FEAT) : h1bf;
        lstm_gemm<<<NT, 128, 0, stream>>>(Ax0, h0bf, Wp0, b_ih0, b_hh0, gates);
        lstm_cell<<<128, 256, 0, stream>>>(gates, c0, h0bf, nullptr);
        lstm_gemm<<<NT, 128, 0, stream>>>(h0bf, h1bf, Wp1, b_ih1, b_hh1, gates);
        float* srow = (s >= WID - 1) ? (seqout + (size_t)(s - (WID - 1)) * BATCH * FEAT)
                                     : nullptr;
        lstm_cell<<<128, 256, 0, stream>>>(gates, c1, h1bf, srow);
    }

    // output conv + concat with input along height
    conv_out_concat<<<32768, 256, 0, stream>>>(x, seqout, w_out, b_out, out);
}
